// AttnReadout_46359876993590
// MI455X (gfx1250) — matrix operand order repaired and checked by reading, not compiled, not run on hardware
//
#include <hip/hip_runtime.h>
#include <math.h>

// Problem constants (match the reference).
#define NN 102400
#define DD 256
#define BB 2048

typedef __attribute__((ext_vector_type(4)))  float  v4f;
typedef __attribute__((ext_vector_type(8)))  float  v8f;
typedef __attribute__((ext_vector_type(16))) __bf16 v16bf;

// ---------------------------------------------------------------------------
// bf16 WMMA: D(16x16,f32) = A(16x32,bf16) * B(32x16,bf16) + C
// ---------------------------------------------------------------------------
__device__ __forceinline__ v8f wmma_bf16(v16bf a, v16bf b, v8f c) {
  return __builtin_amdgcn_wmma_f32_16x16x32_bf16(
      /*neg_a=*/false, a, /*neg_b=*/false, b,
      /*c_mod=*/(short)0, c, /*reuse_a=*/false, /*reuse_b=*/false);
}

// ---------------------------------------------------------------------------
// bf16x3 split-precision GEMM core: one wave computes a 16(M) x 64(N) f32
// tile with K=256, using hi/lo bf16 planes of the weight (pre-split) and
// on-the-fly hi/lo split of the f32 A rows.  Error ~2^-16 relative (al*bl
// term dropped) -- f32-class accuracy on the bf16 matrix pipe.
//
// A operand k order chosen here (16x32, contiguous, not the ISA table order): lane L holds M = L&15;
//   lanes 0-15:  K 0..15 contiguous; lanes 16-31: K 16..31 contiguous.
// B operand k order chosen here (32x16, same as A): lane L holds N = L&15;
//   lanes 0-15:  K 0..15 contiguous; lanes 16-31: K 16..31 contiguous.
// ---------------------------------------------------------------------------
__device__ __forceinline__ void gemm16x64_bf16x3(const float*  __restrict__ arow,
                                                 const __bf16* __restrict__ Wh,
                                                 const __bf16* __restrict__ Wl,
                                                 int lane, int n0, v8f acc[4]) {
  const int nl   = lane & 15;
  const int abase = (lane >> 4) * 16;  // A K sub-offset per lane half
  const int koff  = (lane >> 4) * 16;  // B K sub-offset per lane half

  const __bf16* bh[4];
  const __bf16* bl[4];
#pragma unroll
  for (int t = 0; t < 4; ++t) {
    size_t c = (size_t)(n0 + 16 * t + nl) * DD + koff;
    bh[t] = Wh + c;
    bl[t] = Wl + c;
  }

#pragma unroll 4
  for (int k0 = 0; k0 < DD; k0 += 32) {
    // Load this lane's 16 f32 A elements (one contiguous 16-float chunk).
    const float* ap = arow + k0 + abase;
    v4f a0 = *(const v4f*)(ap);
    v4f a1 = *(const v4f*)(ap + 4);
    v4f a2 = *(const v4f*)(ap + 8);
    v4f a3 = *(const v4f*)(ap + 12);

    // Split into hi/lo bf16 planes, packed per the A operand layout.
    v16bf ah, al;
#pragma unroll
    for (int j = 0; j < 4; ++j) {
      float x0 = a0[j]; __bf16 h0 = (__bf16)x0; ah[j]      = h0; al[j]      = (__bf16)(x0 - (float)h0);
      float x1 = a1[j]; __bf16 h1 = (__bf16)x1; ah[4 + j]  = h1; al[4 + j]  = (__bf16)(x1 - (float)h1);
      float x2 = a2[j]; __bf16 h2 = (__bf16)x2; ah[8 + j]  = h2; al[8 + j]  = (__bf16)(x2 - (float)h2);
      float x3 = a3[j]; __bf16 h3 = (__bf16)x3; ah[12 + j] = h3; al[12 + j] = (__bf16)(x3 - (float)h3);
    }

#pragma unroll
    for (int t = 0; t < 4; ++t) {
      v16bf vbh = *(const v16bf*)(bh[t] + k0);
      v16bf vbl = *(const v16bf*)(bl[t] + k0);
      acc[t] = wmma_bf16(ah, vbh, acc[t]);  // hi*hi
      acc[t] = wmma_bf16(al, vbh, acc[t]);  // lo*hi
      acc[t] = wmma_bf16(ah, vbl, acc[t]);  // hi*lo
    }
  }
}

// ---------------------------------------------------------------------------
// Kernel W: split a [DD,DD] f32 weight into bf16 hi/lo planes.
// ---------------------------------------------------------------------------
__global__ void wsplit_kernel(const float* __restrict__ W,
                              __bf16* __restrict__ hi,
                              __bf16* __restrict__ lo) {
  int i = blockIdx.x * blockDim.x + threadIdx.x;
  float x = W[i];
  __bf16 h = (__bf16)x;
  hi[i] = h;
  lo[i] = (__bf16)(x - (float)h);
}

// ---------------------------------------------------------------------------
// Kernel 0: segment start offsets via binary search (seg_ids is sorted).
// ---------------------------------------------------------------------------
__global__ void seg_bounds_kernel(const int* __restrict__ seg_ids,
                                  int* __restrict__ seg_start) {
  int b = blockIdx.x * blockDim.x + threadIdx.x;
  if (b > BB) return;
  int lo = 0, hi = NN;
  while (lo < hi) {
    int mid = (lo + hi) >> 1;
    if (seg_ids[mid] < b) lo = mid + 1; else hi = mid;
  }
  seg_start[b] = lo;
}

// ---------------------------------------------------------------------------
// Kernel 1: per-segment mean (anchor).  One block per segment, thread = column.
// ---------------------------------------------------------------------------
__global__ void anchor_kernel(const float* __restrict__ ifeat,
                              const int* __restrict__ seg_start,
                              float* __restrict__ anchor_ws,
                              float* __restrict__ out) {
  int b = blockIdx.x;
  int d = threadIdx.x;
  int start = seg_start[b], end = seg_start[b + 1];
  float s = 0.0f;
  for (int i = start; i < end; ++i) s += ifeat[(size_t)i * DD + d];
  float cnt = (float)(end - start);
  if (cnt < 1.0f) cnt = 1.0f;
  float a = s / cnt;
  anchor_ws[(size_t)b * DD + d] = a;
  out[(size_t)b * (2 * DD) + DD + d] = a;
}

// ---------------------------------------------------------------------------
// Kernel 2: feat_v = anchor @ Wv^T + bv.  Grid: BB/16 blocks x 128 threads.
// ---------------------------------------------------------------------------
__global__ void __launch_bounds__(128)
featv_kernel(const float* __restrict__ anchor,
             const __bf16* __restrict__ Wvh,
             const __bf16* __restrict__ Wvl,
             const float* __restrict__ bv,
             float* __restrict__ feat_v) {
  const int m0   = blockIdx.x * 16;
  const int lane = threadIdx.x & 31;
  const int wave = threadIdx.x >> 5;
  const int n0   = wave * 64;
  const int ml   = lane & 15;

  const float* arow = anchor + (size_t)(m0 + ml) * DD;
  v8f acc[4] = {};
  gemm16x64_bf16x3(arow, Wvh, Wvl, lane, n0, acc);

  const int rowadd = (lane >> 4) * 8;  // C layout: VGPR r -> rows r / r+8
#pragma unroll
  for (int t = 0; t < 4; ++t) {
    int col = n0 + t * 16 + ml;
    float bias = bv[col];
#pragma unroll
    for (int r = 0; r < 8; ++r) {
      int row = m0 + r + rowadd;
      feat_v[(size_t)row * DD + col] = acc[t][r] + bias;
    }
  }
}

// ---------------------------------------------------------------------------
// Kernel 3: fused u = ifeat @ Wu^T,  e = sigmoid(u + feat_v[seg]) . we
// Grid: NN/16 blocks x 128 threads (4 waves cover all 256 columns of 16 rows).
// ---------------------------------------------------------------------------
#define SD_STRIDE 264  // padded LDS row stride (floats) to dodge bank conflicts

__global__ void __launch_bounds__(128)
ue_kernel(const float* __restrict__ ifeat,
          const __bf16* __restrict__ Wuh,
          const __bf16* __restrict__ Wul,
          const float* __restrict__ feat_v,
          const float* __restrict__ we,
          const int* __restrict__ seg_ids,
          float* __restrict__ e_out) {
  __shared__ int   sseg[16];
  __shared__ float sdata[16 * SD_STRIDE];
  __shared__ float spart[16 * 8];

  const int m0 = blockIdx.x * 16;
  if (threadIdx.x < 16) sseg[threadIdx.x] = seg_ids[m0 + threadIdx.x];
  __syncthreads();

  const int lane = threadIdx.x & 31;
  const int wave = threadIdx.x >> 5;
  const int n0   = wave * 64;
  const int ml   = lane & 15;

  const float* arow = ifeat + (size_t)(m0 + ml) * DD;
  v8f acc[4] = {};
  gemm16x64_bf16x3(arow, Wuh, Wul, lane, n0, acc);

  // Epilogue: sigmoid(u + feat_v[seg]) * we -> LDS tile.
  const int rowadd = (lane >> 4) * 8;
#pragma unroll
  for (int t = 0; t < 4; ++t) {
    int col = n0 + t * 16 + ml;
    float wc = we[col];
#pragma unroll
    for (int r = 0; r < 8; ++r) {
      int row = r + rowadd;
      float u  = acc[t][r] + feat_v[(size_t)sseg[row] * DD + col];
      float sg = 1.0f / (1.0f + expf(-u));
      sdata[row * SD_STRIDE + col] = sg * wc;
    }
  }
  __syncthreads();

  // Row reduction: 8 threads per row, each sums a stride-8 comb of 32 elements.
  {
    int row = threadIdx.x >> 3;   // 0..15
    int c8  = threadIdx.x & 7;    // 0..7
    const float* p = sdata + row * SD_STRIDE + c8;
    float s = 0.0f;
#pragma unroll
    for (int j = 0; j < 32; ++j) s += p[8 * j];
    spart[row * 8 + c8] = s;
  }
  __syncthreads();
  if (threadIdx.x < 16) {
    float tot = 0.0f;
#pragma unroll
    for (int j = 0; j < 8; ++j) tot += spart[threadIdx.x * 8 + j];
    e_out[m0 + threadIdx.x] = tot;  // TAO == 1.0 -> logits == e
  }
}

// ---------------------------------------------------------------------------
// Kernel 4: per-segment softmax + alpha-weighted sum of ifeat -> out[:, 0:DD].
// ---------------------------------------------------------------------------
__global__ void softmax_readout_kernel(const float* __restrict__ ifeat,
                                       const float* __restrict__ logits,
                                       const int* __restrict__ seg_start,
                                       float* __restrict__ out) {
  __shared__ float red[256];
  __shared__ float alpha[256];
  int b = blockIdx.x;
  int tid = threadIdx.x;
  int start = seg_start[b], end = seg_start[b + 1];

  // segment max
  float m = -3.4e38f;
  for (int i = start + tid; i < end; i += 256) m = fmaxf(m, logits[i]);
  red[tid] = m;
  __syncthreads();
  for (int s = 128; s > 0; s >>= 1) {
    if (tid < s) red[tid] = fmaxf(red[tid], red[tid + s]);
    __syncthreads();
  }
  m = red[0];
  __syncthreads();

  // segment sum of exp
  float z = 0.0f;
  for (int i = start + tid; i < end; i += 256) z += expf(logits[i] - m);
  red[tid] = z;
  __syncthreads();
  for (int s = 128; s > 0; s >>= 1) {
    if (tid < s) red[tid] += red[tid + s];
    __syncthreads();
  }
  float denom = red[0];
  float inv = denom > 0.0f ? 1.0f / denom : 0.0f;

  // weighted sum: alpha staged through LDS in chunks of 256 nodes
  float r = 0.0f;
  for (int base = start; base < end; base += 256) {
    int cnt = end - base;
    if (cnt > 256) cnt = 256;
    __syncthreads();
    if (tid < cnt) alpha[tid] = expf(logits[base + tid] - m) * inv;
    __syncthreads();
    for (int j = 0; j < cnt; ++j)
      r += alpha[j] * ifeat[(size_t)(base + j) * DD + tid];
  }
  out[(size_t)b * (2 * DD) + tid] = r;  // first half: rst
}

// ---------------------------------------------------------------------------
// Launcher
// ---------------------------------------------------------------------------
extern "C" void kernel_launch(void* const* d_in, const int* in_sizes, int n_in,
                              void* d_out, int out_size, void* d_ws, size_t ws_size,
                              hipStream_t stream) {
  const float* ifeat   = (const float*)d_in[0];  // [N, D]
  const float* Wu      = (const float*)d_in[1];  // [D, D]
  const float* Wv      = (const float*)d_in[2];  // [D, D]
  const float* bv      = (const float*)d_in[3];  // [D]
  const float* we      = (const float*)d_in[4];  // [D]
  const int*   seg_ids = (const int*)d_in[5];    // [N]
  float* out = (float*)d_out;                    // [B, 2D]

  char* ws = (char*)d_ws;
  const size_t SEG_OFF   = 0;                      // (B+1) ints, reserve 16 KB
  const size_t ANCH_OFF  = 16384;                  // B*D f32 = 2 MB
  const size_t FV_OFF    = ANCH_OFF + (size_t)BB * DD * 4;
  const size_t LOG_OFF   = FV_OFF   + (size_t)BB * DD * 4;   // N f32, reserve 512 KB
  const size_t WUH_OFF   = LOG_OFF  + 524288;      // DD*DD bf16 = 128 KB each
  const size_t WUL_OFF   = WUH_OFF  + (size_t)DD * DD * 2;
  const size_t WVH_OFF   = WUL_OFF  + (size_t)DD * DD * 2;
  const size_t WVL_OFF   = WVH_OFF  + (size_t)DD * DD * 2;

  int*    seg_start = (int*)(ws + SEG_OFF);
  float*  anchor    = (float*)(ws + ANCH_OFF);
  float*  featv     = (float*)(ws + FV_OFF);
  float*  logits    = (float*)(ws + LOG_OFF);
  __bf16* Wuh       = (__bf16*)(ws + WUH_OFF);
  __bf16* Wul       = (__bf16*)(ws + WUL_OFF);
  __bf16* Wvh       = (__bf16*)(ws + WVH_OFF);
  __bf16* Wvl       = (__bf16*)(ws + WVL_OFF);

  wsplit_kernel<<<DD * DD / 256, 256, 0, stream>>>(Wu, Wuh, Wul);
  wsplit_kernel<<<DD * DD / 256, 256, 0, stream>>>(Wv, Wvh, Wvl);
  seg_bounds_kernel<<<(BB + 1 + 255) / 256, 256, 0, stream>>>(seg_ids, seg_start);
  anchor_kernel<<<BB, 256, 0, stream>>>(ifeat, seg_start, anchor, out);
  featv_kernel<<<BB / 16, 128, 0, stream>>>(anchor, Wvh, Wvl, bv, featv);
  ue_kernel<<<NN / 16, 128, 0, stream>>>(ifeat, Wuh, Wul, featv, we, seg_ids, logits);
  softmax_readout_kernel<<<BB, 256, 0, stream>>>(ifeat, logits, seg_start, out);
}
